// StressTestModel_75892072121038
// MI455X (gfx1250) — matrix operand order repaired and checked by reading, not compiled, not run on hardware
//
#include <hip/hip_runtime.h>
#include <hip/hip_bf16.h>
#include <math.h>

#define PHI_F 1.6180339887498949f

typedef __attribute__((ext_vector_type(16))) __bf16 v16bf;
typedef __attribute__((ext_vector_type(8)))  float  v8f;
typedef __attribute__((ext_vector_type(4)))  unsigned int u32x4;
typedef __attribute__((ext_vector_type(8)))  int i32x8;
typedef __attribute__((ext_vector_type(4)))  int i32x4;

#if __has_builtin(__builtin_amdgcn_tensor_load_to_lds)
#define HAVE_TDM 1
#else
#define HAVE_TDM 0
#endif

// ---------------- K0: W_out f32 -> bf16 ----------------
__global__ void cvt_bf16_kernel(const float* __restrict__ w, __bf16* __restrict__ wb, int n) {
    int i = blockIdx.x * blockDim.x + threadIdx.x;
    if (i < n) wb[i] = (__bf16)w[i];
}

// ---------------- K1: embedding gather + sequential scan ----------------
__global__ void scan_kernel(const int* __restrict__ ids, const float* __restrict__ emb,
                            float* __restrict__ X) {
    int tid = blockIdx.x * blockDim.x + threadIdx.x;   // 0..511
    if (tid >= 512) return;
    int b = tid >> 7;
    int d = tid & 127;
    const int* idb = ids + b * 1024;
    float hr = 0.f, hi = 0.f;
    for (int s = 0; s < 1024; ++s) {
        int id = idb[s];
        float w  = emb[id * 256 + d];
        float bb = emb[id * 256 + 128 + d];
        float wl = 1.f + fabsf(w);
        float th = (hr + hi) / wl + 2.f * (bb + (float)s * PHI_F);
        __sincosf(th, &hi, &hr);                        // hi=sin, hr=cos
        X[(b * 1024 + s) * 128 + d] = hr + hi;
    }
}

// ---------------- K2: q/k sinusoidal features, layout (B,H,S,32) ----------------
__global__ void qk_kernel(const float* __restrict__ X,
                          const float* __restrict__ wq, const float* __restrict__ bq,
                          const float* __restrict__ wk, const float* __restrict__ bk_,
                          float* __restrict__ qb, float* __restrict__ kbuf) {
    int t = blockIdx.x * blockDim.x + threadIdx.x;     // 0..524287
    int dd = t & 15;
    int h  = (t >> 4) & 7;
    int s  = (t >> 7) & 1023;
    int b  = t >> 17;
    float xv = X[(b * 1024 + s) * 128 + h * 16 + dd];
    float tp = (float)s * PHI_F;
    int   wi = h * 16 + dd;
    float thq = xv / (1.f + fabsf(wq[wi])) + bq[wi] + tp;
    float thk = xv / (1.f + fabsf(wk[wi])) + bk_[wi];
    float sq, cq, sk, ck;
    __sincosf(thq, &sq, &cq);
    __sincosf(thk, &sk, &ck);
    int base = ((b * 8 + h) * 1024 + s) * 32 + dd;
    qb[base]        = cq;
    qb[base + 16]   = sq;
    kbuf[base]      = ck;
    kbuf[base + 16] = sk;
}

// ---------------- K3: causal attention, context summed over h and j ----------------
__global__ void attn_kernel(const float* __restrict__ qb, const float* __restrict__ kbuf,
                            const float* __restrict__ X, float* __restrict__ ctx) {
    const int b = blockIdx.y;
    const int i = blockIdx.x;
    const int t = threadIdx.x;
    __shared__ float qrow[8 * 32];
    __shared__ float p[1024];
    __shared__ float red[128];

    for (int u = t; u < 256; u += 128)
        qrow[u] = qb[((b * 8 + (u >> 5)) * 1024 + i) * 32 + (u & 31)];
    __syncthreads();

    float acc = 0.f;
    for (int h = 0; h < 8; ++h) {
        const float* kh = kbuf + (b * 8 + h) * 1024 * 32;
        const float* qh = qrow + h * 32;
        float sj[8];
        float lmax = -INFINITY;
        #pragma unroll
        for (int r = 0; r < 8; ++r) {
            int j = t + r * 128;
            float s = -INFINITY;
            if (j < i) {                                // strict causal (tril k=-1)
                const float* kj = kh + j * 32;
                float d0 = 0.f;
                #pragma unroll
                for (int c = 0; c < 32; ++c) d0 += qh[c] * kj[c];
                s = d0 * 0.17677669529663689f;          // 1/sqrt(2*dh)=1/sqrt(32)
            }
            sj[r] = s;
            lmax = fmaxf(lmax, s);
        }
        red[t] = lmax; __syncthreads();
        for (int o = 64; o > 0; o >>= 1) { if (t < o) red[t] = fmaxf(red[t], red[t + o]); __syncthreads(); }
        float m = red[0]; __syncthreads();

        float lsum = 0.f;
        #pragma unroll
        for (int r = 0; r < 8; ++r) {
            int j = t + r * 128;
            float pv = 0.f;
            if (j < i) pv = __expf(sj[r] - m);
            p[j] = pv;
            lsum += pv;
        }
        red[t] = lsum; __syncthreads();
        for (int o = 64; o > 0; o >>= 1) { if (t < o) red[t] += red[t + o]; __syncthreads(); }
        float l = red[0];
        float invl = (l > 0.f) ? 1.f / l : 0.f;
        __syncthreads();

        const float* Xb = X + b * 1024 * 128 + t;
        float c = 0.f;
        for (int j = 0; j < i; ++j) c += p[j] * Xb[j * 128];
        acc += c * invl;
        __syncthreads();
    }
    ctx[(b * 1024 + i) * 128 + t] = acc;
}

// ---------------- K4: x = X + ctx @ W_ctx^T ----------------
__global__ void ctxproj_kernel(const float* __restrict__ X, const float* __restrict__ ctx,
                               const float* __restrict__ Wctx, float* __restrict__ xbuf) {
    int b = blockIdx.y, i = blockIdx.x, d = threadIdx.x;
    __shared__ float c_sh[128];
    c_sh[d] = ctx[(b * 1024 + i) * 128 + d];
    __syncthreads();
    const float* w = Wctx + d * 128;
    float s = 0.f;
    #pragma unroll 8
    for (int e = 0; e < 128; ++e) s += c_sh[e] * w[e];
    int off = (b * 1024 + i) * 128 + d;
    xbuf[off] = X[off] + s;
}

// ---------------- K5: resonator + gate + SiLU, emit bf16 x2 ----------------
__global__ void resgate_kernel(const float* __restrict__ xbuf,
                               const float* __restrict__ Wres, const float* __restrict__ Bres,
                               const float* __restrict__ Wreal, const float* __restrict__ Wimag,
                               __bf16* __restrict__ x2bf) {
    int b = blockIdx.y, s = blockIdx.x, t = threadIdx.x;
    __shared__ float x_sh[128], cs_sh[128], ss_sh[128];
    x_sh[t] = xbuf[(b * 1024 + s) * 128 + t];
    __syncthreads();
    float tp = (float)s * PHI_F;
    const float* wr = Wres + t * 128;
    const float* br = Bres + t * 128;
    float cs = 0.f, ss = 0.f;
    for (int d = 0; d < 128; ++d) {
        float th = x_sh[d] / (1.f + fabsf(wr[d])) + br[d] + tp;
        float sn, cn;
        __sincosf(th, &sn, &cn);
        cs += cn; ss += sn;
    }
    cs_sh[t] = cs; ss_sh[t] = ss;
    __syncthreads();
    const float* wre = Wreal + t * 128;
    const float* wim = Wimag + t * 128;
    float g = 0.f;
    #pragma unroll 8
    for (int n = 0; n < 128; ++n) g += cs_sh[n] * wre[n] + ss_sh[n] * wim[n];
    float sig = 1.f / (1.f + __expf(-g));
    x2bf[(b * 1024 + s) * 128 + t] = (__bf16)(x_sh[t] + g * sig);
}

// ---------------- TDM helper: 2D tile load Global -> LDS (data_size = 2 bytes) ----------
// D# group0: count=1 | lds_addr | global_addr | type=2.  Group1 per CDNA5 ISA 8.4.
#if HAVE_TDM
__device__ static inline void tdm_load_2d_b16(unsigned int lds_off, const void* gptr,
                                              unsigned int tensor_d0, unsigned int tensor_d1,
                                              unsigned int tile_d0, unsigned int tile_d1,
                                              unsigned int stride_d0) {
    unsigned long long ga = (unsigned long long)(uintptr_t)gptr;
    u32x4 g0;
    g0[0] = 1u;                                            // count=1, user-mode, no gather
    g0[1] = lds_off;                                       // lds_addr (bytes)
    g0[2] = (unsigned int)(ga & 0xFFFFFFFFu);              // global_addr[31:0]
    g0[3] = (unsigned int)((ga >> 32) & 0x1FFFFFFu) | (2u << 30);  // addr[56:32] | type=2
    i32x8 g1;
    g1[0] = (int)(1u << 16);                               // data_size=1 (2 bytes/elem)
    g1[1] = (int)((tensor_d0 & 0xFFFFu) << 16);            // tensor_dim0[15:0] -> bits 63:48
    g1[2] = (int)((tensor_d0 >> 16) | ((tensor_d1 & 0xFFFFu) << 16));
    g1[3] = (int)((tensor_d1 >> 16) | (tile_d0 << 16));    // tensor_dim1[31:16] | tile_dim0
    g1[4] = (int)(tile_d1 & 0xFFFFu);                      // tile_dim1, tile_dim2=0
    g1[5] = (int)stride_d0;                                // tensor_dim0_stride[31:0]
    g1[6] = 0;                                             // stride0[47:32] | stride1[15:0]
    g1[7] = 0;
    i32x4 z4 = {0, 0, 0, 0};
#if defined(__clang_major__) && (__clang_major__ >= 23)
    i32x8 z8 = {0, 0, 0, 0, 0, 0, 0, 0};
    __builtin_amdgcn_tensor_load_to_lds(g0, g1, z4, z4, z8, 0);
#else
    __builtin_amdgcn_tensor_load_to_lds(g0, g1, z4, z4, 0);
#endif
    __builtin_amdgcn_s_wait_tensorcnt(0);
}
#endif

// ---------------- K6: out = x2 @ W_out^T via v_wmma_f32_16x16x32_bf16 ----------------
// Block: 8 waves, one mtile (16 rows). TDM stages the 16x128 bf16 A tile (4 KB) into LDS
// once per block; each wave computes a 16x64 strip (4 n-tiles), reusing the LDS A fragment
// across 4 WMMAs per K-step -> 16 WMMAs per wave.
__global__ void outgemm_kernel(const __bf16* __restrict__ x2bf, const __bf16* __restrict__ wbf,
                               float* __restrict__ out) {
    __shared__ __bf16 Ash[16 * 128];                       // 4 KB A tile
    const int lane  = threadIdx.x & 31;
    const int wave  = threadIdx.x >> 5;
    const int m0    = blockIdx.y * 16;

#if HAVE_TDM
    if (wave == 0) {
        // tensor = x2 (4096 x 128 bf16, row stride 128); tile = rows [m0, m0+16), full K
        tdm_load_2d_b16((unsigned int)(uintptr_t)(&Ash[0]),
                        x2bf + (size_t)m0 * 128,
                        /*tensor_d0=*/128u, /*tensor_d1=*/4096u,
                        /*tile_d0=*/128u,  /*tile_d1=*/16u,
                        /*stride_d0=*/128u);
    }
#else
    for (int u = threadIdx.x; u < 16 * 128; u += 256)
        Ash[u] = x2bf[(size_t)(m0 + (u >> 7)) * 128 + (u & 127)];
#endif
    __syncthreads();

    const int ntbase = blockIdx.x * 32 + wave * 4;         // first of 4 n-tiles for this wave
    if (ntbase >= 2000) return;                            // wave-uniform guard (EXEC stays full)

    const int arow  = lane & 15;
    const int kbA   = (lane & 16) ? 16 : 0;                // A K-half (contiguous, matches B)
    const int koffB = (lane & 16) ? 16 : 0;                // B K-half (contiguous, matches A)
    const int bcol  = lane & 15;

    v8f c0 = {}, c1 = {}, c2 = {}, c3 = {};
    #pragma unroll
    for (int kb = 0; kb < 128; kb += 32) {
        const __bf16* Ap = Ash + arow * 128 + kb + kbA;    // LDS reads (ds_load_b128 x2)
        v16bf a;
        #pragma unroll
        for (int u = 0; u < 16; ++u) { a[u] = Ap[u]; }
        #pragma unroll
        for (int nt = 0; nt < 4; ++nt) {
            const int n0 = (ntbase + nt) * 16;
            const __bf16* Bp = wbf + (size_t)(n0 + bcol) * 128 + kb + koffB;
            v16bf bm;
            #pragma unroll
            for (int u = 0; u < 16; ++u) bm[u] = Bp[u];
            v8f* cp = (nt == 0) ? &c0 : (nt == 1) ? &c1 : (nt == 2) ? &c2 : &c3;
            *cp = __builtin_amdgcn_wmma_f32_16x16x32_bf16(false, a, false, bm, (short)0, *cp,
                                                          false, false);
        }
    }

    #pragma unroll
    for (int nt = 0; nt < 4; ++nt) {
        const int n0 = (ntbase + nt) * 16;
        const v8f c = (nt == 0) ? c0 : (nt == 1) ? c1 : (nt == 2) ? c2 : c3;
        #pragma unroll
        for (int r = 0; r < 8; ++r) {
            int mm = (lane & 16) ? (r + 8) : r;
            out[(size_t)(m0 + mm) * 32000 + n0 + bcol] = c[r];
        }
    }
}

extern "C" void kernel_launch(void* const* d_in, const int* in_sizes, int n_in,
                              void* d_out, int out_size, void* d_ws, size_t ws_size,
                              hipStream_t stream) {
    const int*   input_ids = (const int*)  d_in[0];
    const float* emb       = (const float*)d_in[1];
    const float* w_query   = (const float*)d_in[2];
    const float* b_query   = (const float*)d_in[3];
    const float* w_key     = (const float*)d_in[4];
    const float* b_key     = (const float*)d_in[5];
    const float* W_ctx     = (const float*)d_in[6];
    const float* W_res     = (const float*)d_in[7];
    const float* B_res     = (const float*)d_in[8];
    const float* W_real    = (const float*)d_in[9];
    const float* W_imag    = (const float*)d_in[10];
    const float* W_out     = (const float*)d_in[11];
    float* out = (float*)d_out;

    // Workspace layout (bytes)
    char* ws = (char*)d_ws;
    float*  X     = (float*) (ws + 0);           //  2 MB  (4*1024*128 f32)
    float*  qb    = (float*) (ws + (2u  << 20)); //  4 MB  (4*8*1024*32 f32)
    float*  kbuf  = (float*) (ws + (6u  << 20)); //  4 MB
    float*  ctx   = (float*) (ws + (10u << 20)); //  2 MB
    float*  xbuf  = (float*) (ws + (12u << 20)); //  2 MB
    __bf16* x2bf  = (__bf16*)(ws + (14u << 20)); //  1 MB
    __bf16* wbf   = (__bf16*)(ws + (15u << 20)); //  8 MB (32000*128 bf16)

    cvt_bf16_kernel<<<dim3((4096000 + 255) / 256), dim3(256), 0, stream>>>(W_out, wbf, 4096000);
    scan_kernel<<<dim3(2), dim3(256), 0, stream>>>(input_ids, emb, X);
    qk_kernel<<<dim3(2048), dim3(256), 0, stream>>>(X, w_query, b_query, w_key, b_key, qb, kbuf);
    attn_kernel<<<dim3(1024, 4), dim3(128), 0, stream>>>(qb, kbuf, X, ctx);
    ctxproj_kernel<<<dim3(1024, 4), dim3(128), 0, stream>>>(X, ctx, W_ctx, xbuf);
    resgate_kernel<<<dim3(1024, 4), dim3(128), 0, stream>>>(xbuf, W_res, B_res, W_real, W_imag, x2bf);

    // out = x2 @ W_out^T : M=4096, N=32000, K=128.  2000 n-tiles, 32 per block -> 63 groups.
    outgemm_kernel<<<dim3(63, 256), dim3(256), 0, stream>>>(x2bf, wbf, out);
}
